// SelfAttention_Layer_36232344109173
// MI455X (gfx1250) — 16-bit and 8-bit matrix operand order repaired and checked by reading, compiled, not run on hardware
//
#include <hip/hip_runtime.h>
#include <hip/hip_bf16.h>

typedef int    v16i  __attribute__((ext_vector_type(16)));
typedef int    v8i   __attribute__((ext_vector_type(8)));
typedef int    v4i   __attribute__((ext_vector_type(4)));
typedef unsigned int v4u __attribute__((ext_vector_type(4)));
typedef float  v8f   __attribute__((ext_vector_type(8)));
typedef __bf16 v16bf __attribute__((ext_vector_type(16)));
typedef __bf16 v4bf  __attribute__((ext_vector_type(4)));

#define SB 256
#define SS 512
#define DDIM 256

#define KP_PITCH 272
#define WT_PITCH 272
#define QS_PITCH 272
#define VT_PITCH 80
#define PS_PITCH 80

#define KP_OFF 0
#define WT_OFF (KP_OFF + SS * KP_PITCH)
#define QS_OFF (WT_OFF + DDIM * WT_PITCH)
#define VT_OFF (QS_OFF + 256 * QS_PITCH)
#define PS_OFF (VT_OFF + DDIM * VT_PITCH)
#define MB_OFF (PS_OFF + 256 * PS_PITCH)
#define IF_OFF (MB_OFF + SS * 4)
#define SMEM_BYTES (IF_OFF + DDIM * 4)

#define LSCALE 0.09016843755556021f
#define MASKNEG -6.1963e9f

__device__ __forceinline__ unsigned pk4_fp8(float a, float b, float c, float d) {
  unsigned v = 0;
  v = (unsigned)__builtin_amdgcn_cvt_pk_fp8_f32(a, b, (int)v, false);
  v = (unsigned)__builtin_amdgcn_cvt_pk_fp8_f32(c, d, (int)v, true);
  return v;
}

__device__ __forceinline__ float pe_val(int s, int d, const float* invf) {
  float a = (float)s * invf[d];
  return (d & 1) ? __cosf(a) : __sinf(a);
}

__device__ __forceinline__ void load_bfp8(const unsigned char* row, int g,
                                          v16i& B0, v16i& B1) {
#pragma unroll
  for (int j = 0; j < 4; ++j) {
    uint4 a = *(const uint4*)(row + 32 * j + 16 * g);
    uint4 b = *(const uint4*)(row + 128 + 32 * j + 16 * g);
    B0[4 * j] = (int)a.x; B0[4 * j + 1] = (int)a.y; B0[4 * j + 2] = (int)a.z; B0[4 * j + 3] = (int)a.w;
    B1[4 * j] = (int)b.x; B1[4 * j + 1] = (int)b.y; B1[4 * j + 2] = (int)b.z; B1[4 * j + 3] = (int)b.w;
  }
}

__device__ __forceinline__ v16bf load_bf16_frag(const unsigned char* p0,
                                                const unsigned char* p1) {
  uint4 a = *(const uint4*)p0;
  uint4 b = *(const uint4*)p1;
  v8i t;
  t[0] = (int)a.x; t[1] = (int)a.y; t[2] = (int)a.z; t[3] = (int)a.w;
  t[4] = (int)b.x; t[5] = (int)b.y; t[6] = (int)b.z; t[7] = (int)b.w;
  return __builtin_bit_cast(v16bf, t);
}

__device__ __forceinline__ void tdm_load_v_tile(const float* gsrc, unsigned lds_off) {
  unsigned long long ga = (unsigned long long)(const void*)gsrc;
  v4u g0;
  g0[0] = 1u;
  g0[1] = lds_off;
  g0[2] = (unsigned)(ga & 0xffffffffu);
  g0[3] = (unsigned)((ga >> 32) & 0x01ffffffu)
          | (2u << 30);
  v8i g1;
  g1[0] = (int)(2u << 16);
  g1[1] = (int)(256u << 16);
  g1[2] = (int)(32u << 16);
  g1[3] = (int)(256u << 16);
  g1[4] = 32;
  g1[5] = 256;
  g1[6] = 0;
  g1[7] = 0;
  v4i g2 = {0, 0, 0, 0};
  v4i g3 = {0, 0, 0, 0};
  v8i g4 = {0, 0, 0, 0, 0, 0, 0, 0};
  __builtin_amdgcn_tensor_load_to_lds(g0, g1, g2, g3, g4, 0);
}

__device__ __forceinline__ void project_tile(const float* __restrict__ src,
                                             int rbase, unsigned char* smem,
                                             unsigned char* dst, int dstPitch,
                                             int drow0) {
  const float* invf = (const float*)(smem + IF_OFF);
  const unsigned char* wt = smem + WT_OFF;
  const int lane = threadIdx.x & 31;
  const int m = lane & 15, h = lane >> 4;
  const int srow = rbase + m;
  const float* rp = src + srow * DDIM;

  v16i A0, A1;
#pragma unroll
  for (int p = 0; p < 8; ++p) {
    int kb = 32 * (p >> 1) + 16 * h + 8 * (p & 1);
    float f0[8], f1[8];
#pragma unroll
    for (int t = 0; t < 8; ++t) {
      f0[t] = rp[kb + t] + pe_val(srow, kb + t, invf);
      f1[t] = rp[128 + kb + t] + pe_val(srow, 128 + kb + t, invf);
    }
    A0[2 * p] = (int)pk4_fp8(f0[0], f0[1], f0[2], f0[3]);
    A0[2 * p + 1] = (int)pk4_fp8(f0[4], f0[5], f0[6], f0[7]);
    A1[2 * p] = (int)pk4_fp8(f1[0], f1[1], f1[2], f1[3]);
    A1[2 * p + 1] = (int)pk4_fp8(f1[4], f1[5], f1[6], f1[7]);
  }

#pragma unroll 1
  for (int nb = 0; nb < 16; ++nb) {
    const unsigned char* wr = wt + (nb * 16 + m) * WT_PITCH;
    v16i B0, B1;
    load_bfp8(wr, h, B0, B1);
    v8f acc = {0.f, 0.f, 0.f, 0.f, 0.f, 0.f, 0.f, 0.f};
    acc = __builtin_amdgcn_wmma_f32_16x16x128_fp8_fp8(A0, B0, (short)0, acc, false, false);
    acc = __builtin_amdgcn_wmma_f32_16x16x128_fp8_fp8(A1, B1, (short)0, acc, false, false);
#pragma unroll
    for (int i = 0; i < 8; ++i) {
      float v = acc[i];
      v = v > 0.f ? v : 0.f;
      unsigned pk = (unsigned)__builtin_amdgcn_cvt_pk_fp8_f32(v, v, 0, false);
      dst[(drow0 + 8 * h + i) * dstPitch + nb * 16 + m] = (unsigned char)(pk & 0xffu);
    }
  }
}

__global__ void __launch_bounds__(512) zero_out_kernel(float* p, int n) {
  int i = blockIdx.x * blockDim.x + threadIdx.x;
  if (i < n) p[i] = 0.f;
}

__global__ void __launch_bounds__(512) attn_kernel(
    const float* __restrict__ q, const float* __restrict__ k,
    const float* __restrict__ v, const int* __restrict__ mask,
    const float* __restrict__ W, float* __restrict__ out) {
  __shared__ __align__(16) unsigned char smem[SMEM_BYTES];
  const int b = blockIdx.x;
  const int tid = threadIdx.x;
  const int lane = tid & 31, wv = tid >> 5;
  const int m = lane & 15, h = lane >> 4;

  float* invf = (float*)(smem + IF_OFF);
  float* maskb = (float*)(smem + MB_OFF);
  unsigned char* wt = smem + WT_OFF;
  unsigned char* kp = smem + KP_OFF;
  unsigned char* qs = smem + QS_OFF;
  unsigned char* vt = smem + VT_OFF;
  __bf16* ps = (__bf16*)(smem + PS_OFF + wv * 16 * PS_PITCH);

  if (tid < DDIM)
    invf[tid] = exp2f(-13.287712379549449f * (float)(tid & ~1) * (1.0f / 256.0f));
  maskb[tid] = mask[b * SS + tid] ? 0.0f : MASKNEG;
  for (int idx = tid; idx < DDIM * DDIM; idx += 512) {
    int kd = idx >> 8, e = idx & 255;
    float w = W[idx];
    unsigned pk = (unsigned)__builtin_amdgcn_cvt_pk_fp8_f32(w, w, 0, false);
    wt[e * WT_PITCH + kd] = (unsigned char)(pk & 0xffu);
  }
  __syncthreads();

  const float* qb = q + (size_t)b * SS * DDIM;
  const float* kb2 = k + (size_t)b * SS * DDIM;
  const float* vb = v + (size_t)b * SS * DDIM;

  project_tile(kb2, 32 * wv, smem, kp, KP_PITCH, 32 * wv);
  project_tile(kb2, 32 * wv + 16, smem, kp, KP_PITCH, 32 * wv + 16);
  __syncthreads();

#pragma unroll 1
  for (int pass = 0; pass < 2; ++pass) {
    const int rbase = pass * 256 + wv * 16;
    project_tile(qb, rbase, smem, qs, QS_PITCH, wv * 16);
    __syncthreads();

    v16i QA0, QA1;
    {
      const unsigned char* qr = qs + (wv * 16 + m) * QS_PITCH;
#pragma unroll
      for (int p = 0; p < 8; ++p) {
        int kbo = 32 * (p >> 1) + 16 * h + 8 * (p & 1);
        uint2 u0 = *(const uint2*)(qr + kbo);
        uint2 u1 = *(const uint2*)(qr + 128 + kbo);
        QA0[2 * p] = (int)u0.x; QA0[2 * p + 1] = (int)u0.y;
        QA1[2 * p] = (int)u1.x; QA1[2 * p + 1] = (int)u1.y;
      }
    }
    __syncthreads();

    if (wv == 0) tdm_load_v_tile(vb, QS_OFF);

    v8f O[16];
#pragma unroll
    for (int nb = 0; nb < 16; ++nb)
#pragma unroll
      for (int i = 0; i < 8; ++i) O[nb][i] = 0.f;
    float rm[8], rl[8];
#pragma unroll
    for (int i = 0; i < 8; ++i) { rm[i] = -3.0e38f; rl[i] = 0.f; }

#pragma unroll 1
    for (int kt2 = 0; kt2 < 16; ++kt2) {
      if (wv == 0) __builtin_amdgcn_s_wait_tensorcnt(0);
      __syncthreads();
      if (wv == 0 && kt2 < 15)
        tdm_load_v_tile(vb + (size_t)(kt2 + 1) * 32 * DDIM,
                        QS_OFF + (unsigned)(((kt2 + 1) & 1) * 32768));

      {
        const float* land = (const float*)(smem + QS_OFF + (kt2 & 1) * 32768);
        for (int e = tid; e < 2048; e += 512) {
          int d = e & 255, kg = e >> 8;
          v4bf pk;
          pk[0] = (__bf16)land[(4 * kg + 0) * DDIM + d];
          pk[1] = (__bf16)land[(4 * kg + 1) * DDIM + d];
          pk[2] = (__bf16)land[(4 * kg + 2) * DDIM + d];
          pk[3] = (__bf16)land[(4 * kg + 3) * DDIM + d];
          *(v4bf*)(vt + d * VT_PITCH + kg * 8) = pk;
        }
      }
      __syncthreads();

      v8f L[2];
#pragma unroll
      for (int t = 0; t < 2; ++t) {
        int kt = 2 * kt2 + t;
        const unsigned char* kr = kp + (kt * 16 + m) * KP_PITCH;
        v16i B0, B1;
        load_bfp8(kr, h, B0, B1);
        v8f a = {0.f, 0.f, 0.f, 0.f, 0.f, 0.f, 0.f, 0.f};
        a = __builtin_amdgcn_wmma_f32_16x16x128_fp8_fp8(QA0, B0, (short)0, a, false, false);
        a = __builtin_amdgcn_wmma_f32_16x16x128_fp8_fp8(QA1, B1, (short)0, a, false, false);
        float mb = maskb[kt * 16 + m];
#pragma unroll
        for (int i = 0; i < 8; ++i) a[i] = a[i] * LSCALE + mb;
        L[t] = a;
      }

      float corr[8];
#pragma unroll
      for (int i = 0; i < 8; ++i) {
        float t = fmaxf(L[0][i], L[1][i]);
        t = fmaxf(t, __shfl_xor(t, 1, 16));
        t = fmaxf(t, __shfl_xor(t, 2, 16));
        t = fmaxf(t, __shfl_xor(t, 4, 16));
        t = fmaxf(t, __shfl_xor(t, 8, 16));
        float nm = fmaxf(rm[i], t);
        corr[i] = __builtin_amdgcn_exp2f(rm[i] - nm);
        rm[i] = nm;
        float p0 = __builtin_amdgcn_exp2f(L[0][i] - nm);
        float p1 = __builtin_amdgcn_exp2f(L[1][i] - nm);
        float s = p0 + p1;
        s += __shfl_xor(s, 1, 16);
        s += __shfl_xor(s, 2, 16);
        s += __shfl_xor(s, 4, 16);
        s += __shfl_xor(s, 8, 16);
        rl[i] = rl[i] * corr[i] + s;
        __bf16* pr = ps + (8 * h + i) * (PS_PITCH / 2);
        pr[m] = (__bf16)p0;
        pr[m + 16] = (__bf16)p1;
      }

#pragma unroll
      for (int nb = 0; nb < 16; ++nb)
#pragma unroll
        for (int i = 0; i < 8; ++i) O[nb][i] *= corr[i];

      const unsigned char* prb = (const unsigned char*)ps + m * PS_PITCH;
      v16bf PA = load_bf16_frag(prb + 16 * h, prb + 32 + 16 * h);

#pragma unroll
      for (int nb = 0; nb < 16; ++nb) {
        const unsigned char* vr = vt + (nb * 16 + m) * VT_PITCH + 16 * h;
        v16bf VB = load_bf16_frag(vr, vr + 32);
        O[nb] = __builtin_amdgcn_wmma_f32_16x16x32_bf16(false, PA, false, VB,
                                                        (short)0, O[nb], false, false);
      }
    }

    float rinv[8];
#pragma unroll
    for (int i = 0; i < 8; ++i) rinv[i] = __builtin_amdgcn_rcpf(rl[i]);
#pragma unroll
    for (int nb = 0; nb < 16; ++nb) {
      float cs = 0.f;
#pragma unroll
      for (int i = 0; i < 8; ++i) cs += O[nb][i] * rinv[i];
      cs += __shfl_xor(cs, 16, 32);
      if (h == 0)
        atomicAdd(&out[b * DDIM + nb * 16 + m], cs * (1.0f / 512.0f));
    }
  }
}

extern "C" void kernel_launch(void* const* d_in, const int* in_sizes, int n_in,
                              void* d_out, int out_size, void* d_ws, size_t ws_size,
                              hipStream_t stream) {
  const float* q = (const float*)d_in[0];
  const float* k = (const float*)d_in[1];
  const float* v = (const float*)d_in[2];
  const int* mask = (const int*)d_in[3];
  const float* W = (const float*)d_in[4];
  float* out = (float*)d_out;

  zero_out_kernel<<<(SB * DDIM + 511) / 512, 512, 0, stream>>>(out, SB * DDIM);
  attn_kernel<<<SB, 512, 0, stream>>>(q, k, v, mask, W, out);
}
